// Controller_60601988547087
// MI455X (gfx1250) — matrix operand order repaired and checked by reading, not compiled, not run on hardware
//
#include <hip/hip_runtime.h>

// ENAS LSTM controller on MI455X (gfx1250, wave32).  Single persistent
// 512-thread workgroup: packs weights to f16 WMMA tiles in d_ws, then
// runs the 32-step recurrence with v_wmma_f32_16x16x32_f16 matvecs
// (input vector broadcast across all 16 A rows; weights L2-resident).

#define LL   32
#define DIM  512
#define OPSN 16

typedef __attribute__((ext_vector_type(16))) _Float16     v16h;
typedef __attribute__((ext_vector_type(8)))  float        v8f;
typedef __attribute__((ext_vector_type(8)))  unsigned int v8u;

// workspace layout in dwords
#define WS_WCAT 0
#define WS_WA1  1048576
#define WS_WA2  (WS_WA1 + 131072)
#define WS_EMB  (WS_WA2 + 131072)
#define WS_ANCH (WS_EMB + 4352)
#define WS_AW1  (WS_ANCH + 16896)

__device__ __forceinline__ unsigned pack2h(float a, float b) {
  union { _Float16 h[2]; unsigned u; } cv;
  cv.h[0] = (_Float16)a; cv.h[1] = (_Float16)b;
  return cv.u;
}
__device__ __forceinline__ unsigned rotl32(unsigned v, int n) {
  return (v << n) | (v >> (32 - n));
}
__device__ __noinline__ void threefry2x32(unsigned k0, unsigned k1,
                                          unsigned c0, unsigned c1,
                                          unsigned* o0, unsigned* o1) {
  const unsigned ks2 = 0x1BD11BDAu ^ k0 ^ k1;
  unsigned x0 = c0 + k0, x1 = c1 + k1;
  x0 += x1; x1 = rotl32(x1, 13); x1 ^= x0;
  x0 += x1; x1 = rotl32(x1, 15); x1 ^= x0;
  x0 += x1; x1 = rotl32(x1, 26); x1 ^= x0;
  x0 += x1; x1 = rotl32(x1,  6); x1 ^= x0;
  x0 += k1; x1 += ks2 + 1u;
  x0 += x1; x1 = rotl32(x1, 17); x1 ^= x0;
  x0 += x1; x1 = rotl32(x1, 29); x1 ^= x0;
  x0 += x1; x1 = rotl32(x1, 16); x1 ^= x0;
  x0 += x1; x1 = rotl32(x1, 24); x1 ^= x0;
  x0 += ks2; x1 += k0 + 2u;
  x0 += x1; x1 = rotl32(x1, 13); x1 ^= x0;
  x0 += x1; x1 = rotl32(x1, 15); x1 ^= x0;
  x0 += x1; x1 = rotl32(x1, 26); x1 ^= x0;
  x0 += x1; x1 = rotl32(x1,  6); x1 ^= x0;
  x0 += k0; x1 += k1 + 3u;
  x0 += x1; x1 = rotl32(x1, 17); x1 ^= x0;
  x0 += x1; x1 = rotl32(x1, 29); x1 ^= x0;
  x0 += x1; x1 = rotl32(x1, 16); x1 ^= x0;
  x0 += x1; x1 = rotl32(x1, 24); x1 ^= x0;
  x0 += k1; x1 += ks2 + 4u;
  x0 += x1; x1 = rotl32(x1, 13); x1 ^= x0;
  x0 += x1; x1 = rotl32(x1, 15); x1 ^= x0;
  x0 += x1; x1 = rotl32(x1, 26); x1 ^= x0;
  x0 += x1; x1 = rotl32(x1,  6); x1 ^= x0;
  *o0 = x0 + ks2; *o1 = x1 + k0 + 5u;
}
__device__ __forceinline__ float gumbel_from(unsigned b) {
  union { unsigned u; float f; } cv;
  cv.u = (b >> 9) | 0x3F800000u;
  float u = cv.f - 1.0f;
  u = fmaxf(u, 1.1754944e-38f);
  return -logf(-logf(u));
}
__device__ __forceinline__ float sigm(float x) { return 1.0f / (1.0f + expf(-x)); }

// Matvec via WMMA: A = vector broadcast into all 16 rows (from packed f16
// pairs in LDS), B = pre-swizzled global weight tiles (1KiB each; lane l owns
// 32B; dword v of lane l holds f16 pair W[nt*16+(l&15)][kt*32+2v+16*(l>>4)]).
// B tiles are loaded through an explicit global (AS1) pointer -> global_load,
// and software-pipelined one tile ahead so the pre-WMMA wait is loadcnt<=1.
__device__ __noinline__ void wmma_matvec(const unsigned* tiles, int n_nt, int n_kt,
                                         const unsigned* pk, float* dst, int tid) {
  typedef v8u __attribute__((address_space(1))) gv8u_t;
  const int lane = tid & 31, wave = tid >> 5, grp = lane >> 4, nlo = lane & 15;
#pragma clang loop unroll(disable)
  for (int nt = wave; nt < n_nt; nt += 16) {
    v8f acc = {0.f, 0.f, 0.f, 0.f, 0.f, 0.f, 0.f, 0.f};
    unsigned long long base =
        (unsigned long long)(tiles + nt * n_kt * 256 + lane * 8);
    union { v8u v; unsigned u[8]; v16h h; } af, bcur, bnxt;
    bcur.v = *(const gv8u_t*)base;
    bnxt = bcur;
#pragma clang loop unroll(disable)
    for (int kt = 0; kt < n_kt; ++kt) {
      if (kt + 1 < n_kt)
        bnxt.v = *(const gv8u_t*)(base + (unsigned long long)(kt + 1) * 1024ull);
#pragma unroll
      for (int v = 0; v < 8; ++v) {
        int k0 = 2 * v + 16 * grp;
        af.u[v] = pk[kt * 16 + (k0 >> 1)];
      }
#if __has_builtin(__builtin_amdgcn_wmma_f32_16x16x32_f16)
      acc = __builtin_amdgcn_wmma_f32_16x16x32_f16(
          false, af.h, false, bcur.h, (short)0, acc, false, false);
#else
#pragma unroll
      for (int e = 0; e < 16; ++e) acc[0] += (float)af.h[e] * (float)bcur.h[e];
#endif
      bcur = bnxt;
    }
    if (lane < 16) dst[nt * 16 + nlo] = acc[0];
  }
}

__global__ __launch_bounds__(512)
void enas_controller(const float* emb, const float* w_ih, const float* w_hh,
                     const float* b_ih, const float* b_hh,
                     const float* w_attn_1, const float* w_attn_2,
                     const float* index_fc, const float* op_fc,
                     float* out, unsigned* wsu) {
  __shared__ float    sh_h[DIM], sh_c[DIM], sh_g[4 * DIM], sh_hw[DIM];
  __shared__ unsigned sh_pk[DIM], sh_inp[DIM / 2];
  __shared__ float    sh_lg[LL + 1], sh_ol[OPSN];
  __shared__ int      sh_sel[2];

  const int tid = threadIdx.x;
  unsigned* wcat = wsu + WS_WCAT;
  unsigned* wa1t = wsu + WS_WA1;
  unsigned* wa2t = wsu + WS_WA2;
  unsigned* embp = wsu + WS_EMB;
  float*    anch = (float*)(wsu + WS_ANCH);
  float*    aw1  = (float*)(wsu + WS_AW1);

  // ---- prep: pack weights into WMMA tiles ----
#pragma clang loop unroll(disable)
  for (int i = tid; i < 128 * 32 * 256; i += 512) {
    int tile = i >> 8, w = i & 255;
    int lane = w >> 3, v = w & 7;
    int nt = tile >> 5, kt = tile & 31;
    int n  = nt * 16 + (lane & 15);
    int k0 = kt * 32 + 2 * v + 16 * (lane >> 4);
    float f0, f1;
    if (k0 < DIM) { f0 = w_ih[n * DIM + k0];       f1 = w_ih[n * DIM + k0 + 1]; }
    else          { f0 = w_hh[n * DIM + k0 - DIM]; f1 = w_hh[n * DIM + k0 - DIM + 1]; }
    wcat[i] = pack2h(f0, f1);
  }
#pragma clang loop unroll(disable)
  for (int i = tid; i < 32 * 16 * 256; i += 512) {
    int tile = i >> 8, w = i & 255;
    int lane = w >> 3, v = w & 7;
    int nt = tile >> 4, kt = tile & 15;
    int n  = nt * 16 + (lane & 15);
    int k0 = kt * 32 + 2 * v + 16 * (lane >> 4);
    wa1t[i] = pack2h(w_attn_1[n * DIM + k0], w_attn_1[n * DIM + k0 + 1]);
    wa2t[i] = pack2h(w_attn_2[n * DIM + k0], w_attn_2[n * DIM + k0 + 1]);
  }
#pragma clang loop unroll(disable)
  for (int i = tid; i < 17 * 256; i += 512) {
    int row = i >> 8, p = i & 255;
    embp[i] = pack2h(emb[row * DIM + 2 * p], emb[row * DIM + 2 * p + 1]);
  }
#pragma clang loop unroll(disable)
  for (int i = tid; i < (LL + 1) * DIM; i += 512) anch[i] = 0.0f;
  __threadfence();
  __syncthreads();

  // ---- recurrence ----
  unsigned key0 = 0u, key1 = 42u;       // jax.random.key(42)
  float lp_acc = 0.f, ent_acc = 0.f;
  int node_sel = 0;

  sh_h[tid] = 0.f; sh_c[tid] = 0.f;
  if (tid < 256) { sh_inp[tid] = embp[OPSN * 256 + tid]; }
  __syncthreads();
  if (tid < 256) { sh_pk[tid] = sh_inp[tid]; sh_pk[256 + tid] = 0u; }
  __syncthreads();
  wmma_matvec(wcat, 128, 32, sh_pk, sh_g, tid);
  __syncthreads();
  {
    float gi = sh_g[tid]           + b_ih[tid]           + b_hh[tid];
    float gf = sh_g[DIM + tid]     + b_ih[DIM + tid]     + b_hh[DIM + tid];
    float gg = sh_g[2 * DIM + tid] + b_ih[2 * DIM + tid] + b_hh[2 * DIM + tid];
    float go = sh_g[3 * DIM + tid] + b_ih[3 * DIM + tid] + b_hh[3 * DIM + tid];
    float c2 = sigm(gf) * sh_c[tid] + sigm(gi) * tanhf(gg);
    sh_c[tid] = c2; sh_h[tid] = sigm(go) * tanhf(c2);
  }
  __syncthreads();
  if (tid < 256) sh_pk[tid] = pack2h(sh_h[2 * tid], sh_h[2 * tid + 1]);
  __syncthreads();
  wmma_matvec(wa1t, 32, 16, sh_pk, sh_hw, tid);   // anchors_w1[0] via LDS
  __syncthreads();
  aw1[tid] = sh_hw[tid];
  __threadfence();
  __syncthreads();

#pragma clang loop unroll(disable)
  for (int t = 0; t < LL; ++t) {
    unsigned k1a, k1b, k2a, k2b, nk0, nk1;
    threefry2x32(key0, key1, 0u, 1u, &k1a, &k1b);
    threefry2x32(key0, key1, 2u, 3u, &k2a, &k2b);
    threefry2x32(key0, key1, 4u, 5u, &nk0, &nk1);
    key0 = nk0; key1 = nk1;

    // node cell
    if (tid < 256) {
      sh_pk[tid] = sh_inp[tid];
      sh_pk[256 + tid] = pack2h(sh_h[2 * tid], sh_h[2 * tid + 1]);
    }
    __syncthreads();
    wmma_matvec(wcat, 128, 32, sh_pk, sh_g, tid);
    __syncthreads();
    {
      float gi = sh_g[tid]           + b_ih[tid]           + b_hh[tid];
      float gf = sh_g[DIM + tid]     + b_ih[DIM + tid]     + b_hh[DIM + tid];
      float gg = sh_g[2 * DIM + tid] + b_ih[2 * DIM + tid] + b_hh[2 * DIM + tid];
      float go = sh_g[3 * DIM + tid] + b_ih[3 * DIM + tid] + b_hh[3 * DIM + tid];
      float c2 = sigm(gf) * sh_c[tid] + sigm(gi) * tanhf(gg);
      sh_c[tid] = c2; sh_h[tid] = sigm(go) * tanhf(c2);
    }
    __syncthreads();

    // hw2 = h @ w_attn_2^T
    if (tid < 256) sh_pk[tid] = pack2h(sh_h[2 * tid], sh_h[2 * tid + 1]);
    __syncthreads();
    wmma_matvec(wa2t, 32, 16, sh_pk, sh_hw, tid);
    __syncthreads();

    // idx_logits[a] = 2.5*tanh(dot(tanh(aw1[a]+hw2), index_fc)/5)
    {
      int wv = tid >> 5, lane = tid & 31;
#pragma clang loop unroll(disable)
      for (int a = wv; a <= t; a += 16) {
        const float* awr = aw1 + a * DIM;
        float s = 0.f;
#pragma clang loop unroll(disable)
        for (int i = lane; i < DIM; i += 32)
          s += tanhf(awr[i] + sh_hw[i]) * index_fc[i];
        for (int o = 16; o > 0; o >>= 1) s += __shfl_xor(s, o, 32);
        if (lane == 0) sh_lg[a] = 2.5f * tanhf(s * 0.2f);
      }
    }
    __syncthreads();

    // sample node (thread 0)
    if (tid == 0) {
      int n = t + 1;
      float m = -1e30f;
      float se = 0.f, ent = 0.f;
#pragma clang loop unroll(disable)
      for (int a = 0; a < n; a++) m = fmaxf(m, sh_lg[a]);
#pragma clang loop unroll(disable)
      for (int a = 0; a < n; a++) se += expf(sh_lg[a] - m);
      float lse = logf(se);
#pragma clang loop unroll(disable)
      for (int a = 0; a < n; a++) { float lp = sh_lg[a] - m - lse; ent -= lp * expf(lp); }
      int bi = 0; float best = -1e30f;
#pragma clang loop unroll(disable)
      for (int a = 0; a < n; a++) {
        unsigned r0, r1;
        threefry2x32(k1a, k1b, 0u, (unsigned)a, &r0, &r1);
        float v = sh_lg[a] + gumbel_from(r1);
        if (v > best) { best = v; bi = a; }
      }
      lp_acc  += -(sh_lg[bi] - m - lse);
      ent_acc += ent;
      node_sel = bi;
      sh_sel[0] = bi;
    }
    __syncthreads();
    int nidx = sh_sel[0];

    // op cell: input = anchors[nidx]
    if (tid < 256) {
      const float* ar = anch + nidx * DIM;
      sh_pk[tid]       = pack2h(ar[2 * tid], ar[2 * tid + 1]);
      sh_pk[256 + tid] = pack2h(sh_h[2 * tid], sh_h[2 * tid + 1]);
    }
    __syncthreads();
    wmma_matvec(wcat, 128, 32, sh_pk, sh_g, tid);
    __syncthreads();
    {
      float gi = sh_g[tid]           + b_ih[tid]           + b_hh[tid];
      float gf = sh_g[DIM + tid]     + b_ih[DIM + tid]     + b_hh[DIM + tid];
      float gg = sh_g[2 * DIM + tid] + b_ih[2 * DIM + tid] + b_hh[2 * DIM + tid];
      float go = sh_g[3 * DIM + tid] + b_ih[3 * DIM + tid] + b_hh[3 * DIM + tid];
      float c2 = sigm(gf) * sh_c[tid] + sigm(gi) * tanhf(gg);
      sh_c[tid] = c2; sh_h[tid] = sigm(go) * tanhf(c2);
    }
    __syncthreads();

    // op_logits = tanh((h @ op_fc^T)/5)
    {
      int wv = tid >> 5, lane = tid & 31;
      float s = 0.f;
#pragma clang loop unroll(disable)
      for (int i = lane; i < DIM; i += 32) s += sh_h[i] * op_fc[wv * DIM + i];
      for (int o = 16; o > 0; o >>= 1) s += __shfl_xor(s, o, 32);
      if (lane == 0) sh_ol[wv] = tanhf(s * 0.2f);
    }
    __syncthreads();

    // sample op (thread 0) + emit arch pair
    if (tid == 0) {
      float m = -1e30f;
      float se = 0.f, ent = 0.f;
#pragma clang loop unroll(disable)
      for (int a = 0; a < OPSN; a++) m = fmaxf(m, sh_ol[a]);
#pragma clang loop unroll(disable)
      for (int a = 0; a < OPSN; a++) se += expf(sh_ol[a] - m);
      float lse = logf(se);
#pragma clang loop unroll(disable)
      for (int a = 0; a < OPSN; a++) { float lp = sh_ol[a] - m - lse; ent -= lp * expf(lp); }
      int bi = 0; float best = -1e30f;
#pragma clang loop unroll(disable)
      for (int a = 0; a < OPSN; a++) {
        unsigned r0, r1;
        threefry2x32(k2a, k2b, 0u, (unsigned)a, &r0, &r1);
        float v = sh_ol[a] + gumbel_from(r1);
        if (v > best) { best = v; bi = a; }
      }
      lp_acc  += -(sh_ol[bi] - m - lse);
      ent_acc += ent;
      sh_sel[1] = bi;
      out[2 * t]     = (float)node_sel;
      out[2 * t + 1] = (float)bi;
    }
    __syncthreads();
    int opi = sh_sel[1];

    // inp = emb[op]; anchors[t+1] = h; aw1[t+1] = h @ w_attn_1^T (via LDS)
    if (tid < 256) sh_inp[tid] = embp[opi * 256 + tid];
    anch[(t + 1) * DIM + tid] = sh_h[tid];
    if (tid < 256) sh_pk[tid] = pack2h(sh_h[2 * tid], sh_h[2 * tid + 1]);
    __syncthreads();
    wmma_matvec(wa1t, 32, 16, sh_pk, sh_hw, tid);
    __syncthreads();
    aw1[(t + 1) * DIM + tid] = sh_hw[tid];
    __threadfence();
    __syncthreads();
  }

  if (tid == 0) { out[2 * LL] = ent_acc; out[2 * LL + 1] = lp_acc; }
}

extern "C" void kernel_launch(void* const* d_in, const int* in_sizes, int n_in,
                              void* d_out, int out_size, void* d_ws, size_t ws_size,
                              hipStream_t stream) {
  (void)in_sizes; (void)n_in; (void)out_size; (void)ws_size;
  enas_controller<<<1, 512, 0, stream>>>(
      (const float*)d_in[0], (const float*)d_in[1], (const float*)d_in[2],
      (const float*)d_in[3], (const float*)d_in[4], (const float*)d_in[5],
      (const float*)d_in[6], (const float*)d_in[7], (const float*)d_in[8],
      (float*)d_out, (unsigned*)d_ws);
}
